// DT2W_36094905156165
// MI455X (gfx1250) — compile-verified, matrix operand order checked by reading, not run on hardware
//
#include <hip/hip_runtime.h>

// ---------------------------------------------------------------------------
// Problem constants (from reference setup_inputs):
//   student: (32, 512, 256) f32    teacher: (32, 512, 1024) f32   gamma: int 1
//   output:  scalar f32
// ---------------------------------------------------------------------------
#define BATCH 32
#define NSEQ  512      // N (student length)  == M (teacher length)
#define DCH   256      // reduced channel count
#define CT    1024     // teacher channels
#define PDIAG (NSEQ + NSEQ - 1)   // 1023 anti-diagonals
#define BIGF  1.0e10f

typedef __attribute__((ext_vector_type(16))) __bf16 v16bf;
typedef __attribute__((ext_vector_type(8)))  float  v8f;

// ---------------------------------------------------------------------------
// Kernel 1: teacher channel reduce (mean of 4 consecutive channels) + row norms
// one block per (b, m) row; 256 threads, one output channel each.
// ---------------------------------------------------------------------------
__global__ __launch_bounds__(256) void k_reduce_teacher(
    const float* __restrict__ T, float* __restrict__ Y, float* __restrict__ Y2) {
  const int row = blockIdx.x;          // b*NSEQ + m
  const int c   = threadIdx.x;         // 0..255
  const float4 t = ((const float4*)(T + (size_t)row * CT))[c];
  const float v  = (t.x + t.y + t.z + t.w) * 0.25f;
  Y[(size_t)row * DCH + c] = v;
  __shared__ float red[256];
  red[c] = v * v;
  __syncthreads();
  for (int s = 128; s > 0; s >>= 1) {
    if (c < s) red[c] += red[c + s];
    __syncthreads();
  }
  if (c == 0) Y2[row] = red[0];
}

// ---------------------------------------------------------------------------
// Kernel 2: student row norms ||x_n||^2
// ---------------------------------------------------------------------------
__global__ __launch_bounds__(256) void k_xnorm(
    const float* __restrict__ X, float* __restrict__ X2) {
  const int row = blockIdx.x;
  const int c   = threadIdx.x;
  const float v = X[(size_t)row * DCH + c];
  __shared__ float red[256];
  red[c] = v * v;
  __syncthreads();
  for (int s = 128; s > 0; s >>= 1) {
    if (c < s) red[c] += red[c + s];
    __syncthreads();
  }
  if (c == 0) X2[row] = red[0];
}

// ---------------------------------------------------------------------------
// Kernel 3: D[b,n,m] = x2[n] + y2[m] - 2 * (X Y^T)[n,m] via bf16x3 WMMA.
// One wave computes one 16x16 tile; K loop 256 in steps of 32.
// Split each f32 into hi(bf16) + lo(bf16); product = hi*hi + hi*lo + lo*hi.
// ---------------------------------------------------------------------------
__device__ __forceinline__ void split8(const float4 u, const float4 v, int base,
                                       v16bf& hi, v16bf& lo) {
  const float f[8] = {u.x, u.y, u.z, u.w, v.x, v.y, v.z, v.w};
#pragma unroll
  for (int e = 0; e < 8; ++e) {
    const __bf16 hb = (__bf16)f[e];
    hi[base + e] = hb;
    lo[base + e] = (__bf16)(f[e] - (float)hb);
  }
}

__global__ __launch_bounds__(256) void k_dist_wmma(
    const float* __restrict__ X, const float* __restrict__ Y,
    const float* __restrict__ X2, const float* __restrict__ Y2,
    float* __restrict__ Dout) {
  const int lane = threadIdx.x & 31;
  const int wv   = threadIdx.x >> 5;
  const int w    = blockIdx.x * 8 + wv;      // global wave id, 0..32767
  const int b    = w >> 10;                  // / (32*32 tiles per batch)
  const int t    = w & 1023;
  const int tm   = t >> 5;                   // tile row   (0..31)
  const int tn   = t & 31;                   // tile col   (0..31)
  const int mr   = lane & 15;                // row within tile for this lane
  const int h    = lane >> 4;                // lane half (K striping)

  const float* xrow = X + ((size_t)b * NSEQ + tm * 16 + mr) * DCH;
  const float* yrow = Y + ((size_t)b * NSEQ + tn * 16 + mr) * DCH;

  v8f acc = {};
#pragma unroll
  for (int k0 = 0; k0 < DCH; k0 += 32) {
    // A operand (X rows), documented 16-bit 16x32 layout:
    //   elements 0..7  <- K = k0 + 8h .. +7
    //   elements 8..15 <- K = k0 + 16 + 8h .. +7
    const float4 ax0 = *(const float4*)(xrow + k0 + h * 8);
    const float4 ax1 = *(const float4*)(xrow + k0 + h * 8 + 4);
    const float4 ax2 = *(const float4*)(xrow + k0 + 16 + h * 8);
    const float4 ax3 = *(const float4*)(xrow + k0 + 16 + h * 8 + 4);
    // B operand = Y^T (column-major B == rows of Y, same striping)
    const float4 by0 = *(const float4*)(yrow + k0 + h * 8);
    const float4 by1 = *(const float4*)(yrow + k0 + h * 8 + 4);
    const float4 by2 = *(const float4*)(yrow + k0 + 16 + h * 8);
    const float4 by3 = *(const float4*)(yrow + k0 + 16 + h * 8 + 4);

    v16bf ahi, alo, bhi, blo;
    split8(ax0, ax1, 0, ahi, alo);
    split8(ax2, ax3, 8, ahi, alo);
    split8(by0, by1, 0, bhi, blo);
    split8(by2, by3, 8, bhi, blo);

    acc = __builtin_amdgcn_wmma_f32_16x16x32_bf16(false, ahi, false, bhi,
                                                  (short)0, acc, false, false);
    acc = __builtin_amdgcn_wmma_f32_16x16x32_bf16(false, ahi, false, blo,
                                                  (short)0, acc, false, false);
    acc = __builtin_amdgcn_wmma_f32_16x16x32_bf16(false, alo, false, bhi,
                                                  (short)0, acc, false, false);
  }

  // C/D layout: lane L -> N = L&15, VGPR v -> M = v + 8*(L>>4)
  const int   n  = tn * 16 + (lane & 15);
  const float ys = Y2[b * NSEQ + n];
  float* dbase = Dout + ((size_t)b * NSEQ) * NSEQ;
#pragma unroll
  for (int v = 0; v < 8; ++v) {
    const int gm = tm * 16 + v + 8 * h;
    const float xs = X2[b * NSEQ + gm];
    dbase[(size_t)gm * NSEQ + n] = xs + ys - 2.0f * acc[v];
  }
}

// ---------------------------------------------------------------------------
// Kernel 4: soft-DTW anti-diagonal scan. One block per batch, thread i = row i.
// 3 rotating diagonal rows in LDS; 1023 steps, one barrier each.
// ---------------------------------------------------------------------------
__global__ __launch_bounds__(NSEQ) void k_sdtw_scan(
    const float* __restrict__ D, const int* __restrict__ gptr,
    float* __restrict__ partial) {
  const int b = blockIdx.x;
  const int i = threadIdx.x;                       // 0..511
  const float gamma = (float)(*gptr);
  const float inv_g = 1.0f / gamma;

  __shared__ float buf[3][NSEQ];
  buf[1][i] = BIGF;                                // prev2 at p=0
  buf[2][i] = BIGF;                                // prev  at p=0
  __syncthreads();

  const float* Drow = D + ((size_t)b * NSEQ + i) * NSEQ;
  int cu = 0, pv = 2, pv2 = 1;                     // cur=p%3, prev=(p+2)%3, prev2=(p+1)%3

  for (int p = 0; p < PDIAG; ++p) {
    const int j = p - i;
    const bool valid = (j >= 0) && (j < NSEQ);
    const int jc = j < 0 ? 0 : (j > NSEQ - 1 ? NSEQ - 1 : j);
    const float d = valid ? Drow[jc] : 0.0f;
    if (j + 1 < NSEQ) __builtin_prefetch(Drow + jc + 1, 0, 0);  // next diagonal

    const float a  = (i >= 1) ? buf[pv][i - 1] : BIGF;          // R[i-1, j]
    const float bb = (j >= 1) ? buf[pv][i]     : BIGF;          // R[i, j-1]
    float c = (i >= 1 && j >= 1) ? buf[pv2][i - 1] : BIGF;      // R[i-1, j-1]
    if (i == 0 && j == 0) c = 0.0f;

    const float m = fminf(a, fminf(bb, c));
    const float s = __expf((m - a) * inv_g) + __expf((m - bb) * inv_g) +
                    __expf((m - c) * inv_g);
    const float sm  = m - gamma * __logf(s);
    const float cur = valid ? (d + sm) : BIGF;

    buf[cu][i] = cur;
    __syncthreads();
    const int t0 = pv; pv = cu; cu = pv2; pv2 = t0;             // rotate rows
  }
  // final cur row is now buf[pv]; answer at i = N-1 (j = M-1)
  if (i == NSEQ - 1) partial[b] = buf[pv][NSEQ - 1];
}

// ---------------------------------------------------------------------------
// Kernel 5: deterministic batch mean (no FP atomics).
// ---------------------------------------------------------------------------
__global__ void k_finalize(const float* __restrict__ partial,
                           float* __restrict__ out) {
  if (threadIdx.x == 0) {
    float s = 0.0f;
    for (int b = 0; b < BATCH; ++b) s += partial[b];
    out[0] = s / (float)BATCH;
  }
}

// ---------------------------------------------------------------------------
extern "C" void kernel_launch(void* const* d_in, const int* in_sizes, int n_in,
                              void* d_out, int out_size, void* d_ws, size_t ws_size,
                              hipStream_t stream) {
  const float* student = (const float*)d_in[0];   // (32,512,256) f32
  const float* teacher = (const float*)d_in[1];   // (32,512,1024) f32
  const int*   gamma   = (const int*)d_in[2];     // scalar int 1
  float* out = (float*)d_out;

  // workspace layout (~50.4 MB)
  float* y    = (float*)d_ws;                                   // 32*512*256
  float* Dm   = y  + (size_t)BATCH * NSEQ * DCH;                // 32*512*512
  float* x2   = Dm + (size_t)BATCH * NSEQ * NSEQ;               // 32*512
  float* y2   = x2 + (size_t)BATCH * NSEQ;                      // 32*512
  float* part = y2 + (size_t)BATCH * NSEQ;                      // 32

  (void)in_sizes; (void)n_in; (void)out_size; (void)ws_size;

  // 1) teacher reduce + ||y||^2
  k_reduce_teacher<<<BATCH * NSEQ, 256, 0, stream>>>(teacher, y, y2);
  // 2) ||x||^2
  k_xnorm<<<BATCH * NSEQ, 256, 0, stream>>>(student, x2);
  // 3) pairwise distances via WMMA (32768 tiles, 8 waves/block)
  k_dist_wmma<<<(BATCH * 32 * 32) / 8, 256, 0, stream>>>(student, y, x2, y2, Dm);
  // 4) soft-DTW scan, one WGP-resident block per batch
  k_sdtw_scan<<<BATCH, NSEQ, 0, stream>>>(Dm, gamma, part);
  // 5) batch mean
  k_finalize<<<1, 32, 0, stream>>>(part, out);
}
